// GraphEMALayer_23072564314340
// MI455X (gfx1250) — matrix operand order repaired and checked by reading, not compiled, not run on hardware
//
#include <hip/hip_runtime.h>
#include <hip/hip_bf16.h>

typedef __attribute__((ext_vector_type(8)))  float  v8f;
typedef __attribute__((ext_vector_type(16))) __bf16 v16bf;
typedef __attribute__((ext_vector_type(8)))  __bf16 v8bf;

#define NNODES 50000
#define NEDGES 400000
#define HALFE  200000
#define DIM    128
#define HID    256
#define EPSV   1e-9f

// ---------------- workspace layout ----------------
// float region (float offsets):
//   x2   : [N,128]   @ 0
//   m_A  : [N,128]   @ 6,400,000
//   m_B  : [N,128]   @ 12,800,000
//   deg  : [N]       @ 19,200,000
//   coef : [4*64]    @ 19,250,000      (mag, 1-mag, cos, sin)
//   M    : [E,128]   @ 19,250,256      (float2 aligned)
// bf16 region starts at float offset 70,450,256 (byte offset 281,801,024; 32B aligned):
//   W1t_hi/lo : [256][128] bf16   (transposed, split)
//   W2t_hi/lo : [256][128] bf16
//   Wpt_hi/lo : [128][256] bf16
#define X2_OFF   0
#define MA_OFF   6400000
#define MB_OFF   12800000
#define DEG_OFF  19200000
#define COEF_OFF 19250000
#define M_OFF    19250256
#define BF_OFF   70450256      // float offset where bf16 region begins

// ---------------- tiny helpers ----------------
__global__ void fill_zero_kernel(float* __restrict__ p, int n) {
    int i = blockIdx.x * blockDim.x + threadIdx.x;
    if (i < n) p[i] = 0.0f;
}

// coef[j]=mag, coef[64+j]=1-mag, coef[128+j]=cos(phase), coef[192+j]=sin(phase)
__global__ void coef_kernel(const float* __restrict__ log_dt,
                            const float* __restrict__ llr,
                            const float* __restrict__ lim,
                            float* __restrict__ coef) {
    int j = threadIdx.x;
    if (j < DIM / 2) {
        float dt  = expf(log_dt[0]);
        float mag = expf(-expf(llr[j]) * dt);
        float ph  = lim[j] * dt;
        coef[j]       = mag;
        coef[64 + j]  = 1.0f - mag;
        coef[128 + j] = cosf(ph);
        coef[192 + j] = sinf(ph);
    }
}

// Split fp32 weight W[K][Ncol] (row-major) into transposed bf16 hi/lo: Wt[n][K].
__global__ void pack_w_kernel(const float* __restrict__ W,
                              __bf16* __restrict__ hi, __bf16* __restrict__ lo,
                              int K, int Ncol) {
    int i = blockIdx.x * blockDim.x + threadIdx.x;
    if (i >= K * Ncol) return;
    int k = i / Ncol;
    int n = i - k * Ncol;
    float v = W[i];
    __bf16 h = (__bf16)v;
    hi[n * K + k] = h;
    lo[n * K + k] = (__bf16)(v - (float)h);
}

// Load A operand (16-bit, 16x32 MxK) from an LDS row-major bf16 tile.
// Layout: lane L (m=L%16, h=L/16): elements 0..15 = K[kb+16h .. kb+16h+15].
__device__ __forceinline__ v16bf ldA(const __bf16* base, int m, int kb, int lh, int stride) {
    const v8bf r0 = *(const v8bf*)(base + m * stride + kb + lh * 16);
    const v8bf r1 = *(const v8bf*)(base + m * stride + kb + lh * 16 + 8);
    return __builtin_shufflevector(r0, r1, 0, 1, 2, 3, 4, 5, 6, 7,
                                   8, 9, 10, 11, 12, 13, 14, 15);
}

#define WMMA_BF16(A, B, C) \
    __builtin_amdgcn_wmma_f32_16x16x32_bf16(false, (A), false, (B), (short)0, (C), false, false)

// ---------------- MLP via split-bf16 WMMA (3x 16x16x32 per fp32-accurate step) ----------------
// One workgroup (8 waves) owns a 16-row tile.
// Stage 1: wave w computes hidden col tiles 2w,2w+1 of P1=xW1 and P2=xW2 (K=128),
//          writes silu(P1)*P2 split to bf16 hi/lo in LDS.
// Stage 2: wave w computes output col tile w of H@Wp + bp (K=256).
__global__ __launch_bounds__(256) void mlp_kernel(const float* __restrict__ x,
                                                  const __bf16* __restrict__ W1h,
                                                  const __bf16* __restrict__ W1l,
                                                  const __bf16* __restrict__ W2h,
                                                  const __bf16* __restrict__ W2l,
                                                  const __bf16* __restrict__ Wph,
                                                  const __bf16* __restrict__ Wpl,
                                                  const float* __restrict__ bp,
                                                  float* __restrict__ x2) {
    __shared__ __bf16 sXh[16 * DIM];   // 4 KB
    __shared__ __bf16 sXl[16 * DIM];   // 4 KB
    __shared__ __bf16 sHh[16 * HID];   // 8 KB
    __shared__ __bf16 sHl[16 * HID];   // 8 KB

    const int row0 = blockIdx.x * 16;
    const int t    = threadIdx.x;

    // cooperative load + hi/lo split of the 16x128 x-tile
    for (int i = t; i < 16 * DIM; i += 256) {
        float v = x[row0 * DIM + i];
        __bf16 h = (__bf16)v;
        sXh[i] = h;
        sXl[i] = (__bf16)(v - (float)h);
    }
    __syncthreads();

    const int wave = t >> 5;
    const int lane = t & 31;
    const int lm   = lane & 15;
    const int lh   = lane >> 4;

    // ---- stage 1 ----
    v8f acc1[2] = {};
    v8f acc2[2] = {};
    for (int kb = 0; kb < DIM; kb += 32) {
        const v16bf ah = ldA(sXh, lm, kb, lh, DIM);
        const v16bf al = ldA(sXl, lm, kb, lh, DIM);
#pragma unroll
        for (int cc = 0; cc < 2; ++cc) {
            const int col = (wave * 2 + cc) * 16 + lm;
            const v16bf b1h = *(const v16bf*)(W1h + col * DIM + kb + lh * 16);
            const v16bf b1l = *(const v16bf*)(W1l + col * DIM + kb + lh * 16);
            const v16bf b2h = *(const v16bf*)(W2h + col * DIM + kb + lh * 16);
            const v16bf b2l = *(const v16bf*)(W2l + col * DIM + kb + lh * 16);
            acc1[cc] = WMMA_BF16(ah, b1h, acc1[cc]);
            acc1[cc] = WMMA_BF16(ah, b1l, acc1[cc]);
            acc1[cc] = WMMA_BF16(al, b1h, acc1[cc]);
            acc2[cc] = WMMA_BF16(ah, b2h, acc2[cc]);
            acc2[cc] = WMMA_BF16(ah, b2l, acc2[cc]);
            acc2[cc] = WMMA_BF16(al, b2h, acc2[cc]);
        }
    }
#pragma unroll
    for (int cc = 0; cc < 2; ++cc) {
#pragma unroll
        for (int r = 0; r < 8; ++r) {
            const int m  = r + lh * 8;
            const int nn = (wave * 2 + cc) * 16 + lm;
            const float p1 = acc1[cc][r];
            const float p2 = acc2[cc][r];
            const float hh = (p1 / (1.0f + __expf(-p1))) * p2;   // silu(p1)*p2
            const __bf16 hb = (__bf16)hh;
            sHh[m * HID + nn] = hb;
            sHl[m * HID + nn] = (__bf16)(hh - (float)hb);
        }
    }
    __syncthreads();

    // ---- stage 2 ----
    v8f accp = {};
    const int ocol = wave * 16 + lm;
    for (int kb = 0; kb < HID; kb += 32) {
        const v16bf ah = ldA(sHh, lm, kb, lh, HID);
        const v16bf al = ldA(sHl, lm, kb, lh, HID);
        const v16bf bh = *(const v16bf*)(Wph + ocol * HID + kb + lh * 16);
        const v16bf bl = *(const v16bf*)(Wpl + ocol * HID + kb + lh * 16);
        accp = WMMA_BF16(ah, bh, accp);
        accp = WMMA_BF16(ah, bl, accp);
        accp = WMMA_BF16(al, bh, accp);
    }
    const float bias = bp[ocol];
#pragma unroll
    for (int r = 0; r < 8; ++r) {
        const int m = r + lh * 8;
        x2[(row0 + m) * DIM + ocol] = accp[r] + bias;
    }
}

// ---------------- graph init: M[e]=x2[src], m=segsum(M,dst), deg ----------------
__global__ __launch_bounds__(256) void init_kernel(const int* __restrict__ src,
                                                   const int* __restrict__ dst,
                                                   const float* __restrict__ x2,
                                                   float* __restrict__ M,
                                                   float* __restrict__ m,
                                                   float* __restrict__ deg) {
    const int tid = blockIdx.x * blockDim.x + threadIdx.x;   // < HALFE*64
    const int p = tid >> 6;
    const int j = tid & 63;
    const int a = src[p];
    const int b = dst[p];

    const float2* x2v = (const float2*)x2;
    float2* Mv = (float2*)M;

    const float2 xa = x2v[a * 64 + j];
    const float2 xb = x2v[b * 64 + j];

    Mv[p * 64 + j]           = xa;   // a->b
    Mv[(p + HALFE) * 64 + j] = xb;   // b->a

    atomicAdd(&m[b * DIM + 2 * j],     xa.x);
    atomicAdd(&m[b * DIM + 2 * j + 1], xa.y);
    atomicAdd(&m[a * DIM + 2 * j],     xb.x);
    atomicAdd(&m[a * DIM + 2 * j + 1], xb.y);
    if (j == 0) {
        atomicAdd(&deg[a], 1.0f);
        atomicAdd(&deg[b], 1.0f);
    }
}

// ---------------- one propagation round: in-place pair update + fused scatter ----------------
__global__ __launch_bounds__(256) void msg_kernel(const int* __restrict__ src,
                                                  const int* __restrict__ dst,
                                                  const float* __restrict__ x2,
                                                  const float* __restrict__ deg,
                                                  const float* __restrict__ coef,
                                                  float* __restrict__ M,
                                                  const float* __restrict__ m_cur,
                                                  float* __restrict__ m_next) {
    const int tid = blockIdx.x * blockDim.x + threadIdx.x;   // < HALFE*64
    const int p = tid >> 6;
    const int j = tid & 63;
    const int a = src[p];
    const int b = dst[p];

    const float mag = coef[j];
    const float om  = coef[64 + j];
    const float cc  = coef[128 + j];
    const float ss  = coef[192 + j];

    const float2* x2v = (const float2*)x2;
    const float2* mv  = (const float2*)m_cur;
    float2* Mv = (float2*)M;

    const float2 Mab = Mv[p * 64 + j];
    const float2 Mba = Mv[(p + HALFE) * 64 + j];
    const float2 xa = x2v[a * 64 + j];
    const float2 xb = x2v[b * 64 + j];
    const float2 ma = mv[a * 64 + j];
    const float2 mb = mv[b * 64 + j];
    const float da = deg[a];
    const float db = deg[b];

    float2 na, nb;
    if (da == 1.0f) {
        na = xa;
    } else {
        const float u0 = ma.x - Mba.x;
        const float u1 = ma.y - Mba.y;
        const float inv = 1.0f / (da - 1.0f + EPSV);
        na.x = (cc * xa.x - ss * xa.y) * om + (cc * u0 - ss * u1) * mag * inv;
        na.y = (ss * xa.x + cc * xa.y) * om + (ss * u0 + cc * u1) * mag * inv;
    }
    if (db == 1.0f) {
        nb = xb;
    } else {
        const float u0 = mb.x - Mab.x;
        const float u1 = mb.y - Mab.y;
        const float inv = 1.0f / (db - 1.0f + EPSV);
        nb.x = (cc * xb.x - ss * xb.y) * om + (cc * u0 - ss * u1) * mag * inv;
        nb.y = (ss * xb.x + cc * xb.y) * om + (ss * u0 + cc * u1) * mag * inv;
    }

    Mv[p * 64 + j]           = na;
    Mv[(p + HALFE) * 64 + j] = nb;

    atomicAdd(&m_next[b * DIM + 2 * j],     na.x);
    atomicAdd(&m_next[b * DIM + 2 * j + 1], na.y);
    atomicAdd(&m_next[a * DIM + 2 * j],     nb.x);
    atomicAdd(&m_next[a * DIM + 2 * j + 1], nb.y);
}

// ---------------- final combine ----------------
__global__ __launch_bounds__(256) void final_kernel(const float* __restrict__ x_in,
                                                    const float* __restrict__ x2,
                                                    const float* __restrict__ m,
                                                    const float* __restrict__ deg,
                                                    const float* __restrict__ coef,
                                                    float* __restrict__ out) {
    const int tid = blockIdx.x * blockDim.x + threadIdx.x;   // < N*64
    if (tid >= NNODES * 64) return;
    const int n = tid >> 6;
    const int j = tid & 63;

    const float2* x2v = (const float2*)x2;
    const float2* mv  = (const float2*)m;
    const float2* xiv = (const float2*)x_in;
    float2* ov = (float2*)out;

    const float2 xv = x2v[n * 64 + j];
    const float d = deg[n];
    float2 o;
    if (d == 0.0f) {
        o = xv;
    } else {
        const float mag = coef[j];
        const float om  = coef[64 + j];
        const float cc  = coef[128 + j];
        const float ss  = coef[192 + j];
        const float2 mvv = mv[n * 64 + j];
        const float inv = 1.0f / (d + EPSV);
        o.x = (cc * xv.x - ss * xv.y) * om + (cc * mvv.x - ss * mvv.y) * mag * inv;
        o.y = (ss * xv.x + cc * xv.y) * om + (ss * mvv.x + cc * mvv.y) * mag * inv;
    }
    const float2 xi = xiv[n * 64 + j];
    float2 r;
    r.x = xi.x + fmaxf(o.x, 0.0f);
    r.y = xi.y + fmaxf(o.y, 0.0f);
    ov[n * 64 + j] = r;
}

extern "C" void kernel_launch(void* const* d_in, const int* in_sizes, int n_in,
                              void* d_out, int out_size, void* d_ws, size_t ws_size,
                              hipStream_t stream) {
    const float* x      = (const float*)d_in[0];
    const int*   eidx   = (const int*)d_in[1];
    const float* log_dt = (const float*)d_in[2];
    const float* llr    = (const float*)d_in[3];
    const float* lim    = (const float*)d_in[4];
    const float* W1     = (const float*)d_in[5];
    const float* W2     = (const float*)d_in[6];
    const float* Wp     = (const float*)d_in[7];
    const float* bp     = (const float*)d_in[8];
    float* out = (float*)d_out;

    const int* src = eidx;           // edge_index[0]
    const int* dst = eidx + NEDGES;  // edge_index[1]

    float* ws   = (float*)d_ws;
    float* x2   = ws + X2_OFF;
    float* mA   = ws + MA_OFF;
    float* mB   = ws + MB_OFF;
    float* deg  = ws + DEG_OFF;
    float* coef = ws + COEF_OFF;
    float* M    = ws + M_OFF;

    __bf16* wb  = (__bf16*)(ws + BF_OFF);
    __bf16* W1h = wb;
    __bf16* W1l = wb + 32768;
    __bf16* W2h = wb + 65536;
    __bf16* W2l = wb + 98304;
    __bf16* Wph = wb + 131072;
    __bf16* Wpl = wb + 163840;

    // lambda coefficient table
    coef_kernel<<<1, 64, 0, stream>>>(log_dt, llr, lim, coef);

    // split + transpose weights to bf16 hi/lo (L2-resident afterwards)
    pack_w_kernel<<<(DIM * HID + 255) / 256, 256, 0, stream>>>(W1, W1h, W1l, DIM, HID);
    pack_w_kernel<<<(DIM * HID + 255) / 256, 256, 0, stream>>>(W2, W2h, W2l, DIM, HID);
    pack_w_kernel<<<(HID * DIM + 255) / 256, 256, 0, stream>>>(Wp, Wph, Wpl, HID, DIM);

    // zero deg and m_A
    fill_zero_kernel<<<(NNODES + 255) / 256, 256, 0, stream>>>(deg, NNODES);
    fill_zero_kernel<<<(NNODES * DIM + 255) / 256, 256, 0, stream>>>(mA, NNODES * DIM);

    // MLP: x2 = (silu(xW1) * (xW2)) @ Wp + bp   (split-bf16 WMMA)
    mlp_kernel<<<NNODES / 16, 256, 0, stream>>>(x, W1h, W1l, W2h, W2l, Wph, Wpl, bp, x2);

    // init messages, degrees, first aggregation
    init_kernel<<<(HALFE * 64) / 256, 256, 0, stream>>>(src, dst, x2, M, mA, deg);

    // T = 3 propagation rounds (ping-pong m buffers)
    float* m_cur = mA;
    float* m_next = mB;
    for (int t = 0; t < 3; ++t) {
        fill_zero_kernel<<<(NNODES * DIM + 255) / 256, 256, 0, stream>>>(m_next, NNODES * DIM);
        msg_kernel<<<(HALFE * 64) / 256, 256, 0, stream>>>(src, dst, x2, deg, coef,
                                                           M, m_cur, m_next);
        float* tmp = m_cur; m_cur = m_next; m_next = tmp;
    }

    // final combine
    final_kernel<<<(NNODES * 64 + 255) / 256, 256, 0, stream>>>(x, x2, m_cur, deg, coef, out);
}
